// SGC_class_network_12953621365372
// MI455X (gfx1250) — matrix operand order repaired and checked by reading, not compiled, not run on hardware
//
#include <hip/hip_runtime.h>

#define N_NODES 80000
#define N_EDGES 1280000
#define F_IN    128
#define H3      64
#define GROUPN  40
#define M_POOL  (N_NODES / GROUPN)   // 2000
#define L1D     512
#define L2D     256
#define OUTD    10

// encoded(-inf) under the monotonic float->uint order-preserving map:
// bits(-inf)=0xFF800000 (negative) -> ~bits = 0x007FFFFF
#define ENC_NEG_INF 0x007FFFFFu

typedef _Float16 v16h __attribute__((ext_vector_type(16)));
typedef float    v8f  __attribute__((ext_vector_type(8)));
typedef _Float16 h8   __attribute__((ext_vector_type(8)));

__device__ __forceinline__ float lrelu(float v, float s) { return v > 0.f ? v : s * v; }

// Monotonic float<->uint mapping for atomicMax on floats (deterministic, no CAS loop)
__device__ __forceinline__ unsigned fenc(float f) {
    unsigned u = __float_as_uint(f);
    return (u & 0x80000000u) ? ~u : (u | 0x80000000u);
}
__device__ __forceinline__ float fdec(unsigned e) {
    unsigned u = (e & 0x80000000u) ? (e & 0x7FFFFFFFu) : ~e;
    return __uint_as_float(u);
}

// ---------------- utility fills ----------------
__global__ void fill_f32(float* p, float v, int n) {
    int i = blockIdx.x * blockDim.x + threadIdx.x;
    if (i < n) p[i] = v;
}
__global__ void fill_u32(unsigned* p, unsigned v, int n) {
    int i = blockIdx.x * blockDim.x + threadIdx.x;
    if (i < n) p[i] = v;
}

// ---------------- degree / norm ----------------
__global__ void deg_kernel(const int* __restrict__ dst, float* deg, int E) {
    int e = blockIdx.x * blockDim.x + threadIdx.x;
    if (e < E) atomicAdd(&deg[dst[e]], 1.0f);
}
__global__ void dinv_kernel(const float* __restrict__ deg, float* dinv, int n) {
    int i = blockIdx.x * blockDim.x + threadIdx.x;
    if (i < n) dinv[i] = rsqrtf(deg[i] + 1.0f);  // +1 = self loop
}
__global__ void norm_kernel(const int* __restrict__ src, const int* __restrict__ dst,
                            const float* __restrict__ dinv, float* normE, int E) {
    int e = blockIdx.x * blockDim.x + threadIdx.x;
    if (e < E) normE[e] = dinv[src[e]] * dinv[dst[e]];
}

// ---------------- SGConv propagation: out[dst] += norm * h[src], 128 features ----------
// one 32-lane wave per (edge or self-loop) task; 4 floats per lane
__global__ void prop_kernel(const float* __restrict__ hin, float* __restrict__ hout,
                            const int* __restrict__ src, const int* __restrict__ dst,
                            const float* __restrict__ normE, const float* __restrict__ dinv,
                            int E, int N) {
    long long tid = (long long)blockIdx.x * blockDim.x + threadIdx.x;
    int lane = (int)(tid & 31);
    long long task = tid >> 5;
    if (task >= (long long)E + N) return;
    int s, d; float w;
    if (task < E) { s = src[task]; d = dst[task]; w = normE[task]; }
    else { int i = (int)(task - E); s = i; d = i; float di = dinv[i]; w = di * di; }
    float4 v = ((const float4*)(hin + (size_t)s * F_IN))[lane];
    float* o = hout + (size_t)d * F_IN + lane * 4;
    atomicAdd(o + 0, w * v.x);
    atomicAdd(o + 1, w * v.y);
    atomicAdd(o + 2, w * v.z);
    atomicAdd(o + 3, w * v.w);
}

// ---------------- fp32 -> f16 weight conversion with row padding ----------------
__global__ void wconv_kernel(const float* __restrict__ W, _Float16* __restrict__ Wh,
                             int rows, int cols, int rowsPad) {
    int i = blockIdx.x * blockDim.x + threadIdx.x;
    int tot = rowsPad * cols;
    if (i >= tot) return;
    int r = i / cols;
    Wh[i] = (r < rows) ? (_Float16)W[i] : (_Float16)0.0f;
}

// ---------------- WMMA GEMM: Y = act(A[MxK] * Bh[NoutPad x K]^T + bias) ----------------
// one wave per 16x16 output tile; K-loop in steps of 32 with v_wmma_f32_16x16x32_f16
__global__ void gemm_wmma_f16(const float* __restrict__ A, const _Float16* __restrict__ B,
                              const float* __restrict__ bias, float* __restrict__ Y,
                              int M, int Nout, int NoutPad, int K, int act) {
    int wid  = threadIdx.x >> 5;
    int lane = threadIdx.x & 31;
    int tilesN = NoutPad >> 4;
    int tilesM = M >> 4;
    int tile = blockIdx.x * (blockDim.x >> 5) + wid;
    if (tile >= tilesM * tilesN) return;
    int tm = tile / tilesN, tn = tile % tilesN;
    int n  = lane & 15;
    int hi = lane >> 4;

    const float*    arow = A + (size_t)(tm * 16 + n) * K;  // A row for M = tm*16 + (lane&15)
    const _Float16* brow = B + (size_t)(tn * 16 + n) * K;  // weight row for N = tn*16 + (lane&15)

    v8f c = {};
    for (int kb = 0; kb < K; kb += 32) {
        // A 16x32 f16 layout (ISA 7.12.2): lane holds K = {8*hi..8*hi+7} U {16+8*hi..16+8*hi+7}
        const float4* p0 = (const float4*)(arow + kb + 8 * hi);
        const float4* p1 = (const float4*)(arow + kb + 16 + 8 * hi);
        float4 f0 = p0[0], f1 = p0[1], f2 = p1[0], f3 = p1[1];
        v16h a;
        a[0]  = (_Float16)f0.x; a[1]  = (_Float16)f0.y; a[2]  = (_Float16)f0.z; a[3]  = (_Float16)f0.w;
        a[4]  = (_Float16)f1.x; a[5]  = (_Float16)f1.y; a[6]  = (_Float16)f1.z; a[7]  = (_Float16)f1.w;
        a[8]  = (_Float16)f2.x; a[9]  = (_Float16)f2.y; a[10] = (_Float16)f2.z; a[11] = (_Float16)f2.w;
        a[12] = (_Float16)f3.x; a[13] = (_Float16)f3.y; a[14] = (_Float16)f3.z; a[15] = (_Float16)f3.w;
        // B 32x16 f16 layout: lane (N = lane&15) holds K = {8*hi..8*hi+7} U {16+8*hi..16+8*hi+7}
        const h8* bp = (const h8*)(brow + kb + 8 * hi);
        h8 b0 = bp[0], b1 = bp[2];
        v16h b;
        #pragma unroll
        for (int j = 0; j < 8; ++j) { b[j] = b0[j]; b[8 + j] = b1[j]; }
        c = __builtin_amdgcn_wmma_f32_16x16x32_f16(false, a, false, b, (short)0, c, false, false);
    }

    int col = tn * 16 + n;
    if (col < Nout) {
        float bv = bias ? bias[col] : 0.0f;
        #pragma unroll
        for (int r = 0; r < 8; ++r) {
            int m = tm * 16 + r + 8 * hi;   // C/D layout: VGPR r -> M = r (lanes 0-15), r+8 (16-31)
            float v = c[r] + bv;
            if (act) v = lrelu(v, 0.1f);
            Y[(size_t)m * Nout + col] = v;
        }
    }
}

// ---------------- GAT attention ----------------
__global__ void attn_kernel(const float* __restrict__ g, const float* __restrict__ att_s,
                            const float* __restrict__ att_d, float* a_s, float* a_d, int n) {
    int i = blockIdx.x * blockDim.x + threadIdx.x;
    if (i >= n) return;
    float ss = 0.f, sd = 0.f;
    const float* gr = g + (size_t)i * H3;
    #pragma unroll 4
    for (int f = 0; f < H3; ++f) { float v = gr[f]; ss += v * att_s[f]; sd += v * att_d[f]; }
    a_s[i] = ss; a_d[i] = sd;
}

__global__ void emax_kernel(const int* __restrict__ src, const int* __restrict__ dst,
                            const float* __restrict__ a_s, const float* __restrict__ a_d,
                            unsigned* emaxu, int E, int N) {
    int t = blockIdx.x * blockDim.x + threadIdx.x;
    if (t >= E + N) return;
    int s, d;
    if (t < E) { s = src[t]; d = dst[t]; } else { s = t - E; d = s; }
    float e = lrelu(a_s[s] + a_d[d], 0.2f);
    atomicMax(&emaxu[d], fenc(e));
}

__global__ void ee_kernel(const int* __restrict__ src, const int* __restrict__ dst,
                          const float* __restrict__ a_s, const float* __restrict__ a_d,
                          const unsigned* __restrict__ emaxu, float* eebuf, float* den,
                          int E, int N) {
    int t = blockIdx.x * blockDim.x + threadIdx.x;
    if (t >= E + N) return;
    int s, d;
    if (t < E) { s = src[t]; d = dst[t]; } else { s = t - E; d = s; }
    float e  = lrelu(a_s[s] + a_d[d], 0.2f);
    float ex = expf(e - fdec(emaxu[d]));
    eebuf[t] = ex;
    atomicAdd(&den[d], ex);
}

// one wave per task, 64 features = 32 lanes x float2
__global__ void agg_kernel(const int* __restrict__ src, const int* __restrict__ dst,
                           const float* __restrict__ g, const float* __restrict__ eebuf,
                           const float* __restrict__ den, float* hg, int E, int N) {
    long long tid = (long long)blockIdx.x * blockDim.x + threadIdx.x;
    int lane = (int)(tid & 31);
    long long task = tid >> 5;
    if (task >= (long long)E + N) return;
    int s, d;
    if (task < E) { s = src[task]; d = dst[task]; } else { int i = (int)(task - E); s = i; d = i; }
    float alpha = eebuf[task] / den[d];
    float2 v = ((const float2*)(g + (size_t)s * H3))[lane];
    float* o = hg + (size_t)d * H3 + lane * 2;
    atomicAdd(o + 0, alpha * v.x);
    atomicAdd(o + 1, alpha * v.y);
}

__global__ void biasact_kernel(float* hg, const float* __restrict__ bg, int total) {
    int i = blockIdx.x * blockDim.x + threadIdx.x;
    if (i >= total) return;
    int f = i & (H3 - 1);
    hg[i] = lrelu(hg[i] + bg[f], 0.1f);
}

extern "C" void kernel_launch(void* const* d_in, const int* in_sizes, int n_in,
                              void* d_out, int out_size, void* d_ws, size_t ws_size,
                              hipStream_t stream) {
    const float* x    = (const float*)d_in[0];
    const int*   ei   = (const int*)d_in[1];
    const float* W1   = (const float*)d_in[2];
    const float* b1   = (const float*)d_in[3];
    const float* W2   = (const float*)d_in[4];
    const float* b2   = (const float*)d_in[5];
    const float* Wg   = (const float*)d_in[6];
    const float* atts = (const float*)d_in[7];
    const float* attd = (const float*)d_in[8];
    const float* bg   = (const float*)d_in[9];
    const float* L1   = (const float*)d_in[10];
    const float* bl1  = (const float*)d_in[11];
    const float* L2   = (const float*)d_in[12];
    const float* bl2  = (const float*)d_in[13];
    const float* L3   = (const float*)d_in[14];
    const float* bl3  = (const float*)d_in[15];
    float* out = (float*)d_out;

    const int* srcp = ei;
    const int* dstp = ei + N_EDGES;

    // ---- workspace carve ----
    char* ws = (char*)d_ws;
    size_t off = 0;
    auto carve = [&](size_t bytes) -> char* {
        char* p = ws + off;
        off = (off + bytes + 255) & ~(size_t)255;
        return p;
    };
    float*     h0    = (float*)carve((size_t)N_NODES * F_IN * 4);
    float*     h1    = (float*)carve((size_t)N_NODES * F_IN * 4);
    float*     degb  = (float*)carve((size_t)N_NODES * 4);
    float*     dinv  = (float*)carve((size_t)N_NODES * 4);
    float*     normE = (float*)carve((size_t)N_EDGES * 4);
    float*     g     = (float*)carve((size_t)N_NODES * H3 * 4);
    float*     a_s   = (float*)carve((size_t)N_NODES * 4);
    float*     a_d   = (float*)carve((size_t)N_NODES * 4);
    unsigned*  emaxu = (unsigned*)carve((size_t)N_NODES * 4);
    float*     den   = (float*)carve((size_t)N_NODES * 4);
    float*     eebuf = (float*)carve((size_t)(N_EDGES + N_NODES) * 4);
    _Float16*  W1h   = (_Float16*)carve((size_t)128 * 128 * 2);
    _Float16*  W2h   = (_Float16*)carve((size_t)128 * 128 * 2);
    _Float16*  Wgh   = (_Float16*)carve((size_t)64 * 128 * 2);
    _Float16*  L1h   = (_Float16*)carve((size_t)L1D * (H3 * GROUPN) * 2);
    _Float16*  L2h   = (_Float16*)carve((size_t)L2D * L1D * 2);
    _Float16*  L3h   = (_Float16*)carve((size_t)16 * L2D * 2);
    float* hg   = h1;                       // reuse: h1 free after Wg GEMM
    float* mlp1 = h0;                       // reuse: h0 free after Wg GEMM
    float* mlp2 = h0 + (size_t)M_POOL * L1D;

    const int B = 256;
    auto cdiv = [](long long a, long long b) { return (int)((a + b - 1) / b); };
    long long etasks = (long long)(N_EDGES + N_NODES);
    int nf = N_NODES * F_IN;

    // ---- degrees / norms ----
    fill_f32<<<cdiv(N_NODES, B), B, 0, stream>>>(degb, 0.f, N_NODES);
    deg_kernel<<<cdiv(N_EDGES, B), B, 0, stream>>>(dstp, degb, N_EDGES);
    dinv_kernel<<<cdiv(N_NODES, B), B, 0, stream>>>(degb, dinv, N_NODES);
    norm_kernel<<<cdiv(N_EDGES, B), B, 0, stream>>>(srcp, dstp, dinv, normE, N_EDGES);

    // ---- weights -> f16 (row-padded) ----
    wconv_kernel<<<cdiv(128 * 128, B), B, 0, stream>>>(W1, W1h, 128, 128, 128);
    wconv_kernel<<<cdiv(128 * 128, B), B, 0, stream>>>(W2, W2h, 128, 128, 128);
    wconv_kernel<<<cdiv(64 * 128, B), B, 0, stream>>>(Wg, Wgh, 64, 128, 64);
    wconv_kernel<<<cdiv((long long)L1D * H3 * GROUPN, B), B, 0, stream>>>(L1, L1h, L1D, H3 * GROUPN, L1D);
    wconv_kernel<<<cdiv(L2D * L1D, B), B, 0, stream>>>(L2, L2h, L2D, L1D, L2D);
    wconv_kernel<<<cdiv(16 * L2D, B), B, 0, stream>>>(L3, L3h, OUTD, L2D, 16);

    // ---- SGConv 1: prop, prop, linear W1 ----
    fill_f32<<<cdiv(nf, B), B, 0, stream>>>(h1, 0.f, nf);
    prop_kernel<<<cdiv(etasks * 32, B), B, 0, stream>>>(x, h1, srcp, dstp, normE, dinv, N_EDGES, N_NODES);
    fill_f32<<<cdiv(nf, B), B, 0, stream>>>(h0, 0.f, nf);
    prop_kernel<<<cdiv(etasks * 32, B), B, 0, stream>>>(h1, h0, srcp, dstp, normE, dinv, N_EDGES, N_NODES);
    gemm_wmma_f16<<<cdiv((N_NODES / 16) * (128 / 16), 8), B, 0, stream>>>(h0, W1h, b1, h1, N_NODES, 128, 128, 128, 1);

    // ---- SGConv 2: prop, prop, linear W2 ----
    fill_f32<<<cdiv(nf, B), B, 0, stream>>>(h0, 0.f, nf);
    prop_kernel<<<cdiv(etasks * 32, B), B, 0, stream>>>(h1, h0, srcp, dstp, normE, dinv, N_EDGES, N_NODES);
    fill_f32<<<cdiv(nf, B), B, 0, stream>>>(h1, 0.f, nf);
    prop_kernel<<<cdiv(etasks * 32, B), B, 0, stream>>>(h0, h1, srcp, dstp, normE, dinv, N_EDGES, N_NODES);
    gemm_wmma_f16<<<cdiv((N_NODES / 16) * (128 / 16), 8), B, 0, stream>>>(h1, W2h, b2, h0, N_NODES, 128, 128, 128, 1);

    // ---- GAT: g = h @ Wg^T (no bias/act yet) ----
    gemm_wmma_f16<<<cdiv((N_NODES / 16) * (H3 / 16), 8), B, 0, stream>>>(h0, Wgh, nullptr, g, N_NODES, H3, H3, 128, 0);
    attn_kernel<<<cdiv(N_NODES, B), B, 0, stream>>>(g, atts, attd, a_s, a_d, N_NODES);
    fill_u32<<<cdiv(N_NODES, B), B, 0, stream>>>(emaxu, ENC_NEG_INF, N_NODES); // enc(-inf)
    emax_kernel<<<cdiv(etasks, B), B, 0, stream>>>(srcp, dstp, a_s, a_d, emaxu, N_EDGES, N_NODES);
    fill_f32<<<cdiv(N_NODES, B), B, 0, stream>>>(den, 0.f, N_NODES);
    ee_kernel<<<cdiv(etasks, B), B, 0, stream>>>(srcp, dstp, a_s, a_d, emaxu, eebuf, den, N_EDGES, N_NODES);
    fill_f32<<<cdiv(N_NODES * H3, B), B, 0, stream>>>(hg, 0.f, N_NODES * H3);
    agg_kernel<<<cdiv(etasks * 32, B), B, 0, stream>>>(srcp, dstp, g, eebuf, den, hg, N_EDGES, N_NODES);
    biasact_kernel<<<cdiv(N_NODES * H3, B), B, 0, stream>>>(hg, bg, N_NODES * H3);

    // ---- pooled MLP head: [2000 x 2560] -> 512 -> 256 -> 10 ----
    gemm_wmma_f16<<<cdiv((M_POOL / 16) * (L1D / 16), 8), B, 0, stream>>>(hg, L1h, bl1, mlp1, M_POOL, L1D, L1D, H3 * GROUPN, 1);
    gemm_wmma_f16<<<cdiv((M_POOL / 16) * (L2D / 16), 8), B, 0, stream>>>(mlp1, L2h, bl2, mlp2, M_POOL, L2D, L2D, L1D, 1);
    gemm_wmma_f16<<<cdiv((M_POOL / 16) * (16 / 16), 8), B, 0, stream>>>(mlp2, L3h, bl3, out, M_POOL, OUTD, 16, L2D, 1);

    (void)in_sizes; (void)n_in; (void)out_size; (void)ws_size;
}
